// CrossAtt_64063732187816
// MI455X (gfx1250) — compile-verified, matrix operand order checked by reading, not run on hardware
//
#include <hip/hip_runtime.h>

#define DMODEL 768
#define NHEAD  12
#define DHEAD  64
#define SEQ_Q  2048
#define SEQ_K  2048
#define NBATCH 4
#define NTOK   (NBATCH * SEQ_Q)

typedef __bf16 bf16_t;
typedef float  v8f   __attribute__((ext_vector_type(8)));
typedef bf16_t v16bf __attribute__((ext_vector_type(16)));
typedef bf16_t v8bf  __attribute__((ext_vector_type(8)));

union F16U { v16bf v; v8bf h[2]; bf16_t e[16]; };
union F8FU { v8f v; float e[8]; };
union B8U  { v8bf v; bf16_t e[8]; };

#if __has_builtin(__builtin_amdgcn_tensor_load_to_lds) && \
    __has_builtin(__builtin_amdgcn_s_wait_tensorcnt)
#define HAVE_TDM 1
typedef unsigned int u32x4 __attribute__((ext_vector_type(4)));
typedef int          i32x8 __attribute__((ext_vector_type(8)));
typedef int          i32x4 __attribute__((ext_vector_type(4)));
#else
#define HAVE_TDM 0
#endif

__device__ __forceinline__ v16bf frag_from(const bf16_t* p0, const bf16_t* p1) {
  F16U f;
  f.h[0] = *(const v8bf*)p0;
  f.h[1] = *(const v8bf*)p1;
  return f.v;
}

__device__ __forceinline__ v8f wmma_bf16(v16bf a, v16bf b, v8f c) {
  return __builtin_amdgcn_wmma_f32_16x16x32_bf16(false, a, false, b, (short)0, c,
                                                 false, false);
}

// ---------------------------------------------------------------------------
// valid length per batch row
// ---------------------------------------------------------------------------
__global__ __launch_bounds__(256) void vlen_kernel(const int* __restrict__ mask,
                                                   int* __restrict__ vlen) {
  __shared__ int red[256];
  const int b = blockIdx.x;
  const int t = threadIdx.x;
  int c = 0;
  for (int i = t; i < SEQ_K; i += 256) c += (mask[b * SEQ_K + i] != 0);
  red[t] = c;
  __syncthreads();
  for (int off = 128; off > 0; off >>= 1) {
    if (t < off) red[t] += red[t + off];
    __syncthreads();
  }
  if (t == 0) vlen[b] = red[0];
}

// ---------------------------------------------------------------------------
// Generic bf16-WMMA GEMM: C[M,N] = act( A[M,K] @ W[K,N] + bias ) + addend
// A either f32 (Af, converted during LDS staging) or bf16 (Abf, staged by the
// Tensor Data Mover when available). Block: 256 threads (8 waves), tile
// 128x128, wave sub-tile 32x64 (8 x v_wmma_f32_16x16x32_bf16 per K-step).
// ---------------------------------------------------------------------------
__global__ __launch_bounds__(256) void gemm_kernel(
    const float* __restrict__ Af, const bf16_t* __restrict__ Abf,
    const float* __restrict__ W, const float* __restrict__ bias,
    const float* __restrict__ addend, float* __restrict__ Cf,
    bf16_t* __restrict__ Cbf, int M, int N, int K, int relu) {
  constexpr int LDT = 40;  // bf16 stride: 32 payload + 8 pad, 80B rows
  __shared__ bf16_t As[128 * LDT];
  __shared__ bf16_t Bs[128 * LDT];

  const int tid = threadIdx.x;
  const int lane = tid & 31;
  const int wid = tid >> 5;
  const int wm = wid & 3;    // 4 wave-rows of 32
  const int wn = wid >> 2;   // 2 wave-cols of 64
  const int half = lane >> 4;
  const int ln = lane & 15;
  const int kb = half ? 8 : 0;
  const int mbase = blockIdx.y * 128;
  const int nbase = blockIdx.x * 128;

  v8f acc[2][4];
#pragma unroll
  for (int mi = 0; mi < 2; ++mi)
#pragma unroll
    for (int ni = 0; ni < 4; ++ni) acc[mi][ni] = (v8f)0.0f;

  for (int k0 = 0; k0 < K; k0 += 32) {
    // ---- stage A tile: 128 rows x 32 k ----
    if (Abf) {
#if HAVE_TDM
      // Tensor Data Mover: one DMA for the whole 128x32 bf16 tile, with LDS
      // padding (pad every 16 DWORDs by 4 DWORDs) reproducing stride LDT=40.
      if (wid == 0) {
        const bf16_t* gsrc = Abf + (size_t)mbase * K + k0;
        const unsigned long long ga = (unsigned long long)(uintptr_t)gsrc;
        const unsigned int lds_off = (unsigned int)(uintptr_t)(&As[0]);
        u32x4 g0;
        g0[0] = 1u;  // count=1, user descriptor
        g0[1] = lds_off;
        g0[2] = (unsigned int)(ga & 0xFFFFFFFFu);
        g0[3] = (unsigned int)(ga >> 32) | (2u << 30);  // type=2 ("image")
        i32x8 g1;
        // data_size=2B, pad_enable, pad_interval=16 DW (code 3),
        // pad_amount=4 DW (code 3)
        g1[0] = (int)((1u << 16) | (1u << 20) | (3u << 22) | (3u << 25));
        g1[1] = (int)(768u << 16);        // tensor_dim0[15:0] in [31:16]
        g1[2] = 0;                        // tensor_dim0 hi / tensor_dim1 lo
        g1[3] = (int)(16u | (32u << 16)); // tensor_dim1 = 1<<20, tile_dim0 = 32
        g1[4] = 128;                      // tile_dim1 = 128, tile_dim2 = 0
        g1[5] = 768;                      // tensor_dim0_stride = 768
        g1[6] = 0;
        g1[7] = 0;
        i32x4 z4 = {0, 0, 0, 0};
        i32x8 z8 = {0, 0, 0, 0, 0, 0, 0, 0};
        __builtin_amdgcn_tensor_load_to_lds(g0, g1, z4, z4, z8, 0);
      }
#else
      {
        const int row = tid >> 1;
        const int koff = (tid & 1) * 16;
        bf16_t* dst = &As[row * LDT + koff];
        const bf16_t* src = Abf + (size_t)(mbase + row) * K + k0 + koff;
        *(v8bf*)dst = *(const v8bf*)src;
        *(v8bf*)(dst + 8) = *(const v8bf*)(src + 8);
      }
#endif
    } else {
      const int row = tid >> 1;
      const int koff = (tid & 1) * 16;
      bf16_t* dst = &As[row * LDT + koff];
      const float4* src =
          (const float4*)(Af + (size_t)(mbase + row) * K + k0 + koff);
      float4 x0 = src[0], x1 = src[1], x2 = src[2], x3 = src[3];
      B8U c0, c1;
      c0.e[0] = (bf16_t)x0.x; c0.e[1] = (bf16_t)x0.y;
      c0.e[2] = (bf16_t)x0.z; c0.e[3] = (bf16_t)x0.w;
      c0.e[4] = (bf16_t)x1.x; c0.e[5] = (bf16_t)x1.y;
      c0.e[6] = (bf16_t)x1.z; c0.e[7] = (bf16_t)x1.w;
      c1.e[0] = (bf16_t)x2.x; c1.e[1] = (bf16_t)x2.y;
      c1.e[2] = (bf16_t)x2.z; c1.e[3] = (bf16_t)x2.w;
      c1.e[4] = (bf16_t)x3.x; c1.e[5] = (bf16_t)x3.y;
      c1.e[6] = (bf16_t)x3.z; c1.e[7] = (bf16_t)x3.w;
      *(v8bf*)dst = c0.v;
      *(v8bf*)(dst + 8) = c1.v;
    }
    // ---- stage B tile transposed: Bs[n][k], 32 k x 128 n ----
    {
      const int kk = tid >> 3;          // 0..31
      const int noff = (tid & 7) * 16;  // 0..112
      const float4* src =
          (const float4*)(W + (size_t)(k0 + kk) * N + nbase + noff);
      float4 x0 = src[0], x1 = src[1], x2 = src[2], x3 = src[3];
      float tmp[16] = {x0.x, x0.y, x0.z, x0.w, x1.x, x1.y, x1.z, x1.w,
                       x2.x, x2.y, x2.z, x2.w, x3.x, x3.y, x3.z, x3.w};
#pragma unroll
      for (int j = 0; j < 16; ++j)
        Bs[(noff + j) * LDT + kk] = (bf16_t)tmp[j];
    }
    // ---- prefetch next K-tile while this one is consumed ----
    if (k0 + 32 < K) {
      if (!Abf)
        __builtin_prefetch(
            Af + (size_t)(mbase + (tid >> 1)) * K + k0 + 32 + (tid & 1) * 16, 0,
            3);
      __builtin_prefetch(
          W + (size_t)(k0 + 32 + (tid >> 3)) * N + nbase + (tid & 7) * 16, 0,
          3);
    }
#if HAVE_TDM
    if (Abf && wid == 0) __builtin_amdgcn_s_wait_tensorcnt(0);
#endif
    __syncthreads();

    v16bf afrag[2], bfrag[4];
#pragma unroll
    for (int mi = 0; mi < 2; ++mi) {
      const bf16_t* p = &As[(wm * 32 + mi * 16 + ln) * LDT];
      afrag[mi] = frag_from(p + kb, p + 16 + kb);
    }
#pragma unroll
    for (int ni = 0; ni < 4; ++ni) {
      const bf16_t* p = &Bs[(wn * 64 + ni * 16 + ln) * LDT];
      bfrag[ni] = frag_from(p + kb, p + 16 + kb);
    }
#pragma unroll
    for (int mi = 0; mi < 2; ++mi)
#pragma unroll
      for (int ni = 0; ni < 4; ++ni)
        acc[mi][ni] = wmma_bf16(afrag[mi], bfrag[ni], acc[mi][ni]);
    __syncthreads();
  }

  // ---- epilogue ----
#pragma unroll
  for (int mi = 0; mi < 2; ++mi) {
#pragma unroll
    for (int ni = 0; ni < 4; ++ni) {
      const int col = nbase + wn * 64 + ni * 16 + ln;
      const float bv = bias ? bias[col] : 0.0f;
      F8FU u;
      u.v = acc[mi][ni];
#pragma unroll
      for (int r = 0; r < 8; ++r) {
        const int row = mbase + wm * 32 + mi * 16 + r + half * 8;
        float val = u.e[r] + bv;
        if (relu) val = fmaxf(val, 0.0f);
        if (addend) val += addend[(size_t)row * N + col];
        if (Cf) Cf[(size_t)row * N + col] = val;
        if (Cbf) Cbf[(size_t)row * N + col] = (bf16_t)val;
      }
    }
  }
}

// ---------------------------------------------------------------------------
// Fused flash attention. Block = 128 threads (4 waves), each wave = one 16-row
// Q tile; block covers 64 q rows of one (batch, head). Loops keys in 32-chunks.
// ---------------------------------------------------------------------------
__global__ __launch_bounds__(128) void attn_kernel(
    const bf16_t* __restrict__ q, const bf16_t* __restrict__ k,
    const bf16_t* __restrict__ v, const int* __restrict__ vlen,
    float* __restrict__ ctx) {
  __shared__ bf16_t vt[DHEAD * 32];       // V tile transposed [d][key]
  __shared__ bf16_t pbuf[4][16 * 32];     // per-wave P repack buffer

  const int tid = threadIdx.x;
  const int lane = tid & 31;
  const int wid = tid >> 5;
  const int half = lane >> 4;
  const int ln = lane & 15;
  const int kb = half ? 8 : 0;
  const int b = blockIdx.z;
  const int h = blockIdx.y;
  const int hd = h * DHEAD;
  const int q0 = blockIdx.x * 64 + wid * 16;  // q-row base for this wave
  const int vl = vlen[b];
  const float scale = 0.125f;  // 1/sqrt(64)

  // Q fragments (16 rows x 64 dh = two K=32 slices), loaded once from global
  const bf16_t* qp = q + (size_t)(b * SEQ_Q + q0 + ln) * DMODEL + hd;
  v16bf qf[2];
  qf[0] = frag_from(qp + kb, qp + 16 + kb);
  qf[1] = frag_from(qp + 32 + kb, qp + 48 + kb);

  v8f o[4];
#pragma unroll
  for (int t4 = 0; t4 < 4; ++t4) o[t4] = (v8f)0.0f;
  float mrow[8], lrow[8];
#pragma unroll
  for (int r = 0; r < 8; ++r) { mrow[r] = -1e30f; lrow[r] = 0.0f; }

  for (int kt = 0; kt < SEQ_K; kt += 32) {
    __syncthreads();  // vt safe to overwrite
    // ---- cooperatively stage V tile transposed: vt[d][key] ----
    {
      const int kk = tid >> 2;  // 0..31 key
      const int dg = tid & 3;   // 0..3 group of 16 d
      const bf16_t* src =
          v + (size_t)(b * SEQ_K + kt + kk) * DMODEL + hd + dg * 16;
      B8U x0, x1;
      x0.v = *(const v8bf*)src;
      x1.v = *(const v8bf*)(src + 8);
#pragma unroll
      for (int j = 0; j < 8; ++j) {
        vt[(dg * 16 + j) * 32 + kk] = x0.e[j];
        vt[(dg * 16 + 8 + j) * 32 + kk] = x1.e[j];
      }
    }

    // ---- scores for 32 keys = two 16-key WMMA tiles ----
    float sv[2][8];
#pragma unroll
    for (int s = 0; s < 2; ++s) {
      const int key = kt + s * 16 + ln;  // this lane's key column
      const bf16_t* kp = k + (size_t)(b * SEQ_K + key) * DMODEL + hd;
      v16bf kf0 = frag_from(kp + kb, kp + 16 + kb);
      v16bf kf1 = frag_from(kp + 32 + kb, kp + 48 + kb);
      v8f sc = (v8f)0.0f;
      sc = wmma_bf16(qf[0], kf0, sc);
      sc = wmma_bf16(qf[1], kf1, sc);
      F8FU u;
      u.v = sc;
      const bool ok = key < vl;
#pragma unroll
      for (int r = 0; r < 8; ++r) sv[s][r] = ok ? u.e[r] * scale : -1e6f;
    }

    // ---- online softmax (row stats reduced across 16-lane halves) ----
    float pv[2][8];
#pragma unroll
    for (int r = 0; r < 8; ++r) {
      float mx = fmaxf(sv[0][r], sv[1][r]);
#pragma unroll
      for (int off = 1; off < 16; off <<= 1)
        mx = fmaxf(mx, __shfl_xor(mx, off, 16));
      const float mnew = fmaxf(mrow[r], mx);
      const float corr = __expf(mrow[r] - mnew);
      mrow[r] = mnew;
      float rs = 0.0f;
#pragma unroll
      for (int s = 0; s < 2; ++s) {
        const float e = __expf(sv[s][r] - mnew);
        pv[s][r] = e;
        rs += e;
      }
#pragma unroll
      for (int off = 1; off < 16; off <<= 1) rs += __shfl_xor(rs, off, 16);
      lrow[r] = lrow[r] * corr + rs;
#pragma unroll
      for (int t4 = 0; t4 < 4; ++t4) {
        F8FU u;
        u.v = o[t4];
        u.e[r] *= corr;
        o[t4] = u.v;
      }
    }

    // ---- repack P (C-layout -> A-fragment layout) via LDS ----
    bf16_t* pb = pbuf[wid];
#pragma unroll
    for (int s = 0; s < 2; ++s)
#pragma unroll
      for (int r = 0; r < 8; ++r)
        pb[(r + half * 8) * 32 + s * 16 + ln] = (bf16_t)pv[s][r];
    __syncthreads();  // vt staged + pbuf visible

    const bf16_t* pp = pb + ln * 32;
    v16bf pf = frag_from(pp + kb, pp + 16 + kb);
#pragma unroll
    for (int t4 = 0; t4 < 4; ++t4) {
      const bf16_t* vp = &vt[(t4 * 16 + ln) * 32];
      v16bf vf = frag_from(vp + kb, vp + 16 + kb);
      o[t4] = wmma_bf16(pf, vf, o[t4]);
    }
  }

  // ---- normalize and write context (f32, head-concat layout) ----
#pragma unroll
  for (int t4 = 0; t4 < 4; ++t4) {
    F8FU u;
    u.v = o[t4];
#pragma unroll
    for (int r = 0; r < 8; ++r) {
      const int tok = b * SEQ_Q + q0 + r + half * 8;
      ctx[(size_t)tok * DMODEL + hd + t4 * 16 + ln] = u.e[r] / lrow[r];
    }
  }
}

// ---------------------------------------------------------------------------
// Row-wise LayerNorm: one block per token row
// ---------------------------------------------------------------------------
__global__ __launch_bounds__(256) void ln_kernel(const float* __restrict__ y,
                                                 const float* __restrict__ g,
                                                 const float* __restrict__ be,
                                                 float* __restrict__ out) {
  __shared__ float s1[256], s2[256];
  const int row = blockIdx.x;
  const int t = threadIdx.x;
  const float* yr = y + (size_t)row * DMODEL;
  float a = 0.0f, bq = 0.0f;
  for (int i = t; i < DMODEL; i += 256) {
    const float x = yr[i];
    a += x;
    bq += x * x;
  }
  s1[t] = a;
  s2[t] = bq;
  __syncthreads();
  for (int off = 128; off > 0; off >>= 1) {
    if (t < off) {
      s1[t] += s1[t + off];
      s2[t] += s2[t + off];
    }
    __syncthreads();
  }
  const float mu = s1[0] / (float)DMODEL;
  const float var = s2[0] / (float)DMODEL - mu * mu;
  const float inv = rsqrtf(var + 1e-5f);
  for (int i = t; i < DMODEL; i += 256)
    out[(size_t)row * DMODEL + i] = (yr[i] - mu) * inv * g[i] + be[i];
}

// ---------------------------------------------------------------------------
extern "C" void kernel_launch(void* const* d_in, const int* in_sizes, int n_in,
                              void* d_out, int out_size, void* d_ws,
                              size_t ws_size, hipStream_t stream) {
  const float* queries = (const float*)d_in[0];
  const float* keys = (const float*)d_in[1];
  const float* values = (const float*)d_in[2];
  const int* mask = (const int*)d_in[3];
  const float* Wq = (const float*)d_in[4];
  const float* Wk = (const float*)d_in[5];
  const float* Wv = (const float*)d_in[6];
  const float* Wo = (const float*)d_in[7];
  const float* W1 = (const float*)d_in[8];
  const float* b1 = (const float*)d_in[9];
  const float* W2 = (const float*)d_in[10];
  const float* b2 = (const float*)d_in[11];
  const float* ln_g = (const float*)d_in[12];
  const float* ln_b = (const float*)d_in[13];
  float* out = (float*)d_out;

  char* ws = (char*)d_ws;
  const size_t SZ = (size_t)NTOK * DMODEL;  // 6,291,456 elements
  bf16_t* q_bf = (bf16_t*)ws;
  bf16_t* k_bf = q_bf + SZ;
  bf16_t* v_bf = k_bf + SZ;
  bf16_t* h1_bf = v_bf + SZ;
  float* ctx_f = (float*)(ws + 8 * SZ);
  float* att_f = ctx_f + SZ;
  float* y_f = att_f + SZ;
  int* vlen = (int*)(ws + 20 * SZ);

  dim3 gg(DMODEL / 128, NTOK / 128);  // (6, 64)
  dim3 gb(256);

  vlen_kernel<<<NBATCH, 256, 0, stream>>>(mask, vlen);

  gemm_kernel<<<gg, gb, 0, stream>>>(queries, nullptr, Wq, nullptr, nullptr,
                                     nullptr, q_bf, NTOK, DMODEL, DMODEL, 0);
  gemm_kernel<<<gg, gb, 0, stream>>>(keys, nullptr, Wk, nullptr, nullptr,
                                     nullptr, k_bf, NTOK, DMODEL, DMODEL, 0);
  gemm_kernel<<<gg, gb, 0, stream>>>(values, nullptr, Wv, nullptr, nullptr,
                                     nullptr, v_bf, NTOK, DMODEL, DMODEL, 0);

  attn_kernel<<<dim3(SEQ_Q / 64, NHEAD, NBATCH), 128, 0, stream>>>(
      q_bf, k_bf, v_bf, vlen, ctx_f);

  // att_out = ctx @ Wo (f32 for residual)
  gemm_kernel<<<gg, gb, 0, stream>>>(ctx_f, nullptr, Wo, nullptr, nullptr,
                                     att_f, nullptr, NTOK, DMODEL, DMODEL, 0);
  // h1 = relu(att_out @ W1 + b1) (bf16 for next GEMM)
  gemm_kernel<<<gg, gb, 0, stream>>>(att_f, nullptr, W1, b1, nullptr, nullptr,
                                     h1_bf, NTOK, DMODEL, DMODEL, 1);
  // y = h1 @ W2 + b2 + att_out   (A tile staged via TDM when available)
  gemm_kernel<<<gg, gb, 0, stream>>>(nullptr, h1_bf, W2, b2, att_f, y_f,
                                     nullptr, NTOK, DMODEL, DMODEL, 0);

  ln_kernel<<<NTOK, 256, 0, stream>>>(y_f, ln_g, ln_b, out);
}
